// BackBone8x_16355235463873
// MI455X (gfx1250) — matrix operand order repaired and checked by reading, not compiled, not run on hardware
//
#include <hip/hip_runtime.h>

typedef unsigned short u16;
typedef __attribute__((ext_vector_type(16))) __bf16 v16bf;
typedef __attribute__((ext_vector_type(8)))  float  v8f;
typedef __attribute__((ext_vector_type(8)))  unsigned short us8;
typedef __attribute__((ext_vector_type(4)))  unsigned short us4;

// ---------- bf16 helpers (raw-bit storage; RNE float->bf16) ----------
__device__ __forceinline__ u16 f2bf(float f) {
    unsigned u = __builtin_bit_cast(unsigned, f);
    unsigned r = (u + 0x7FFFu + ((u >> 16) & 1u)) >> 16;
    return (u16)r;
}
__device__ __forceinline__ float bf2f(u16 h) {
    unsigned u = ((unsigned)h) << 16;
    return __builtin_bit_cast(float, u);
}
__device__ __forceinline__ __bf16 bfraw(u16 h) { return __builtin_bit_cast(__bf16, h); }

// 16-element bf16 fragment from two 16B loads (global or LDS pointers).
__device__ __forceinline__ v16bf load_pair16(const u16* p0, const u16* p1) {
    us8 lo = *(const us8*)p0;
    us8 hi = *(const us8*)p1;
    v16bf r;
#pragma unroll
    for (int e = 0; e < 8; ++e) { r[e] = bfraw(lo[e]); r[e + 8] = bfraw(hi[e]); }
    return r;
}
// 16-element bf16 fragment from four 8B loads.
__device__ __forceinline__ v16bf load_quad4(const u16* p0, const u16* p1,
                                            const u16* p2, const u16* p3) {
    us4 q0 = *(const us4*)p0;
    us4 q1 = *(const us4*)p1;
    us4 q2 = *(const us4*)p2;
    us4 q3 = *(const us4*)p3;
    v16bf r;
#pragma unroll
    for (int e = 0; e < 4; ++e) {
        r[e]      = bfraw(q0[e]);
        r[e + 4]  = bfraw(q1[e]);
        r[e + 8]  = bfraw(q2[e]);
        r[e + 12] = bfraw(q3[e]);
    }
    return r;
}

// Branchless row offset: all bounds combined with bitwise &, u32 cndmask select.
// OOB redirects to a zeroed tail region at element offset `zoff`.
__device__ __forceinline__ unsigned row_off(int mv, int z, int y, int x,
                                            int Din, int Hin, int Win,
                                            int cin, unsigned zoff) {
    int ok = mv & (int)(z >= 0) & (int)(z < Din)
                & (int)(y >= 0) & (int)(y < Hin)
                & (int)(x >= 0) & (int)(x < Win);
    unsigned idx = (unsigned)((z * Hin + y) * Win + x) * (unsigned)cin;
    return ok ? idx : zoff;
}

__device__ __forceinline__ v8f wmma_bf16(v16bf a, v16bf b, v8f c) {
    return __builtin_amdgcn_wmma_f32_16x16x32_bf16(false, a, false, b, (short)0, c, false, false);
}

// ---------- utility fills ----------
__global__ void fill_u16(u16* p, size_t n) {
    size_t i  = (size_t)blockIdx.x * blockDim.x + threadIdx.x;
    size_t st = (size_t)gridDim.x * blockDim.x;
    for (; i < n; i += st) p[i] = 0;
}
__global__ void fill_u8(unsigned char* p, size_t n) {
    size_t i  = (size_t)blockIdx.x * blockDim.x + threadIdx.x;
    size_t st = (size_t)gridDim.x * blockDim.x;
    for (; i < n; i += st) p[i] = 0;
}

// ---------- scatter voxels into dense bf16 grid (channel-last) ----------
__global__ void scatter_voxels(const float* __restrict__ vf, const int* __restrict__ coords,
                               u16* __restrict__ grid, unsigned char* __restrict__ mask,
                               int n, int H, int W, int C) {
    int i = blockIdx.x * blockDim.x + threadIdx.x;
    if (i >= n) return;
    int z = coords[i * 4 + 1], y = coords[i * 4 + 2], x = coords[i * 4 + 3];
    size_t cell = (size_t)(z * H + y) * W + x;
    size_t base = cell * C;
    for (int c = 0; c < C; ++c) grid[base + c] = f2bf(vf[i * C + c]);
    mask[cell] = 1;
}

// ---------- weight repack (cout,cin,kt) fp32 -> (t,cout,cin) bf16 ----------
__global__ void wprep(const float* __restrict__ w, u16* __restrict__ wb,
                      int kt, int cout, int cin) {
    int i = blockIdx.x * blockDim.x + threadIdx.x;
    int tot = kt * cout * cin;
    if (i >= tot) return;
    int t = i / (cout * cin);
    int r = i % (cout * cin);
    int n = r / cin, k = r % cin;
    wb[(size_t)(t * cout + n) * cin + k] = f2bf(w[((size_t)n * cin + k) * kt + t]);
}

// ---------- fold BN: scale = g*rsqrt(v+eps); bias = b - m*scale ----------
__global__ void bnprep(const float* __restrict__ g, const float* __restrict__ b,
                       const float* __restrict__ m, const float* __restrict__ v,
                       float* __restrict__ bn, int cout) {
    int n = blockIdx.x * blockDim.x + threadIdx.x;
    if (n >= cout) return;
    float sc = g[n] * rsqrtf(v[n] + 1e-3f);
    bn[n]       = sc;
    bn[128 + n] = b[n] - m[n] * sc;
}

// ---------- mask dilation for spconv layers ----------
__global__ void mask_update(const unsigned char* __restrict__ mi, unsigned char* __restrict__ mo,
                            int Din, int Hin, int Win, int Dout, int Hout, int Wout,
                            int kd, int kh, int kw, int sd, int sh, int sw,
                            int pd, int ph, int pw) {
    int v = blockIdx.x * blockDim.x + threadIdx.x;
    int Mt = Dout * Hout * Wout;
    if (v >= Mt) return;
    int od = v / (Hout * Wout);
    int r  = v % (Hout * Wout);
    int oh = r / Wout, ow = r % Wout;
    unsigned char acc = 0;
    for (int zt = 0; zt < kd; ++zt) {
        int z = od * sd - pd + zt;
        if (z < 0 || z >= Din) continue;
        for (int yt = 0; yt < kh; ++yt) {
            int y = oh * sh - ph + yt;
            if (y < 0 || y >= Hin) continue;
            for (int xt = 0; xt < kw; ++xt) {
                int x = ow * sw - pw + xt;
                if (x < 0 || x >= Win) continue;
                acc |= mi[(size_t)(z * Hin + y) * Win + x];
            }
        }
    }
    mo[v] = acc ? 1 : 0;
}

// Shared prologue/epilogue pieces for the conv kernels ---------------------
// Part 1: wave/tile ids + optional LDS weight staging (barrier-safe), then exit.
#define CONV_HEAD(CIN_)                                                           \
    extern __shared__ u16 smem[];                                                 \
    const int lane = threadIdx.x & 31;                                            \
    const int wave = blockIdx.x * (blockDim.x >> 5) + (threadIdx.x >> 5);         \
    const int Mtot = Dout * Hout * Wout;                                          \
    const int tilesN = cout >> 4;                                                 \
    const int tilesM = (Mtot + 31) >> 5;                                          \
    const bool validw = wave < tilesM * tilesN;                                   \
    if (STAGE) {                                                                  \
        for (unsigned i = (unsigned)threadIdx.x * 8; i < wel; i += (unsigned)blockDim.x * 8) { \
            *(us8*)(smem + i) = *(const us8*)(wb + i);                            \
        }                                                                         \
        __syncthreads();                                                          \
    }                                                                             \
    if (!validw) return;                                                          \
    const u16* __restrict__ wsrc = STAGE ? (const u16*)smem : wb;                 \
    const int tm = wave / tilesN;                                                 \
    const int tn = wave % tilesN;                                                 \
    const int half = lane >> 4;                                                   \
    const int ml   = lane & 15;                                                   \
    const int vm0 = tm * 32 + ml;                                                 \
    const int vm1 = vm0 + 16;                                                     \
    const int mv0 = vm0 < Mtot;                                                   \
    const int mv1 = vm1 < Mtot;                                                   \
    const int vc0 = mv0 ? vm0 : Mtot - 1;  /* clamped decode, branchless */       \
    const int vc1 = mv1 ? vm1 : Mtot - 1;                                         \
    const int od0 = vc0 / (Hout * Wout), rr0 = vc0 % (Hout * Wout);               \
    const int oh0 = rr0 / Wout, ow0 = rr0 % Wout;                                 \
    const int od1 = vc1 / (Hout * Wout), rr1 = vc1 % (Hout * Wout);               \
    const int oh1 = rr1 / Wout, ow1 = rr1 % Wout;                                 \
    const int iz0 = od0 * sd - pd, iy0 = oh0 * sh - ph, ix0 = ow0 * sw - pw;      \
    const int iz1 = od1 * sd - pd, iy1 = oh1 * sh - ph, ix1 = ow1 * sw - pw;      \
    const unsigned wbase = (unsigned)(tn * 16 + ml) * (unsigned)(CIN_);           \
    const unsigned wts = (unsigned)cout * (unsigned)(CIN_);                       \
    __builtin_prefetch(in, 0, 1);                                                 \
    v8f acc0 = {};                                                                \
    v8f acc1 = {};

#define CONV_STORE(ACC_, VROW_)                                                   \
    {                                                                             \
        float v_ = (ACC_) * sc + bi;                                              \
        v_ = v_ > 0.f ? v_ : 0.f;                                                 \
        v_ *= (float)omask[(VROW_)];                                              \
        out[(size_t)(VROW_) * cout + n] = f2bf(v_);                               \
    }

#define CONV_EPILOGUE()                                                           \
    const int n = tn * 16 + ml;                                                   \
    const float sc = bn[n];                                                       \
    const float bi = bn[128 + n];                                                 \
    if (tm * 32 + 32 <= Mtot) {                                                   \
        _Pragma("unroll")                                                         \
        for (int r = 0; r < 8; ++r) CONV_STORE(acc0[r], tm * 32 + r + 8 * half)   \
        _Pragma("unroll")                                                         \
        for (int r = 0; r < 8; ++r) CONV_STORE(acc1[r], tm * 32 + 16 + r + 8 * half) \
    } else {                                                                      \
        _Pragma("unroll")                                                         \
        for (int r = 0; r < 8; ++r) {                                             \
            int vrow = tm * 32 + r + 8 * half;                                    \
            if (vrow < Mtot) CONV_STORE(acc0[r], vrow)                            \
        }                                                                         \
        _Pragma("unroll")                                                         \
        for (int r = 0; r < 8; ++r) {                                             \
            int vrow = tm * 32 + 16 + r + 8 * half;                               \
            if (vrow < Mtot) CONV_STORE(acc1[r], vrow)                            \
        }                                                                         \
    }

// ---------- conv, cin >= 32 (full 32-deep K chunks per tap) ----------
template <int CIN, int KD, int KH, int KW, bool STAGE>
__global__ void conv_big(const u16* __restrict__ in, u16* __restrict__ out,
                         const u16* __restrict__ wb, const float* __restrict__ bn,
                         const unsigned char* __restrict__ omask,
                         unsigned zoff_in, unsigned wel,
                         int Din, int Hin, int Win, int Dout, int Hout, int Wout,
                         int cout, int sd, int sh, int sw, int pd, int ph, int pw) {
    CONV_HEAD(CIN)
    int t = 0;
    for (int zt = 0; zt < KD; ++zt) {
        const int z0 = iz0 + zt, z1 = iz1 + zt;
        for (int yt = 0; yt < KH; ++yt) {
            const int y0 = iy0 + yt, y1 = iy1 + yt;
#pragma unroll
            for (int xt = 0; xt < KW; ++xt, ++t) {
                const u16* row0 = in + row_off(mv0, z0, y0, ix0 + xt, Din, Hin, Win, CIN, zoff_in);
                const u16* row1 = in + row_off(mv1, z1, y1, ix1 + xt, Din, Hin, Win, CIN, zoff_in);
                const u16* wrow = wsrc + (wbase + (unsigned)t * wts);
#pragma unroll
                for (int c0 = 0; c0 < CIN; c0 += 32) {
                    const u16* p0 = row0 + c0;
                    const u16* p1 = row1 + c0;
                    v16bf a0 = load_pair16(p0 + 8 * half, p0 + 16 + 8 * half);
                    v16bf a1 = load_pair16(p1 + 8 * half, p1 + 16 + 8 * half);
                    v16bf b  = load_pair16(wrow + c0 + 8 * half, wrow + c0 + 16 + 8 * half);
                    acc0 = wmma_bf16(a0, b, acc0);
                    acc1 = wmma_bf16(a1, b, acc1);
                }
            }
        }
    }
    CONV_EPILOGUE()
}

// ---------- conv, cin == 16: two taps packed per 32-deep K chunk ----------
template <int KD, int KH, int KW, bool STAGE>
__global__ void conv_c16(const u16* __restrict__ in, u16* __restrict__ out,
                         const u16* __restrict__ wb, const float* __restrict__ bn,
                         const unsigned char* __restrict__ omask,
                         unsigned zoff_in, unsigned zoff_w, unsigned wel,
                         int Din, int Hin, int Win, int Dout, int Hout, int Wout,
                         int cout, int sd, int sh, int sw, int pd, int ph, int pw) {
    constexpr int CIN = 16;
    constexpr int KT  = KD * KH * KW;
    CONV_HEAD(CIN)
#pragma unroll
    for (int t = 0; t < KT; t += 2) {
        const int zA = t / (KH * KW), yA = (t / KW) % KH, xA = t % KW;
        const int t1 = t + 1;
        const bool hasB = t1 < KT;   // compile-time per unrolled iteration
        const int zB = hasB ? t1 / (KH * KW) : 0;
        const int yB = hasB ? (t1 / KW) % KH : 0;
        const int xB = hasB ? t1 % KW : 0;

        const u16* rA0 = in + row_off(mv0, iz0 + zA, iy0 + yA, ix0 + xA, Din, Hin, Win, CIN, zoff_in);
        const u16* rA1 = in + row_off(mv1, iz1 + zA, iy1 + yA, ix1 + xA, Din, Hin, Win, CIN, zoff_in);
        const u16* rB0 = in + (hasB ? row_off(mv0, iz0 + zB, iy0 + yB, ix0 + xB, Din, Hin, Win, CIN, zoff_in) : zoff_in);
        const u16* rB1 = in + (hasB ? row_off(mv1, iz1 + zB, iy1 + yB, ix1 + xB, Din, Hin, Win, CIN, zoff_in) : zoff_in);
        const unsigned wAo = wbase + (unsigned)t * wts;
        const unsigned wBo = hasB ? wbase + (unsigned)t1 * wts : zoff_w;

        // A: K half h = channels 8h..8h+7 of tap A (K 0..15) and tap B (K 16..31)
        v16bf a0 = load_pair16(rA0 + 8 * half, rB0 + 8 * half);
        v16bf a1 = load_pair16(rA1 + 8 * half, rB1 + 8 * half);
        // B: K half h = weights 8h..8h+7 of tap A (K 0..15) and tap B (K 16..31)
        v16bf b = load_pair16(wsrc + wAo + 8 * half, wsrc + wBo + 8 * half);
        acc0 = wmma_bf16(a0, b, acc0);
        acc1 = wmma_bf16(a1, b, acc1);
    }
    CONV_EPILOGUE()
}

// ---------- conv, cin == 4: eight taps packed per 32-deep K chunk ----------
template <int KD, int KH, int KW, bool STAGE>
__global__ void conv_c4(const u16* __restrict__ in, u16* __restrict__ out,
                        const u16* __restrict__ wb, const float* __restrict__ bn,
                        const unsigned char* __restrict__ omask,
                        unsigned zoff_in, unsigned zoff_w, unsigned wel,
                        int Din, int Hin, int Win, int Dout, int Hout, int Wout,
                        int cout, int sd, int sh, int sw, int pd, int ph, int pw) {
    constexpr int CIN = 4;
    constexpr int KT  = KD * KH * KW;
    CONV_HEAD(CIN)
#pragma unroll
    for (int t0 = 0; t0 < KT; t0 += 8) {
        unsigned r0o[8], r1o[8], wro[8];
#pragma unroll
        for (int j = 0; j < 8; ++j) {
            const int t = t0 + j;
            if (t < KT) {
                const int zt = t / (KH * KW), yt = (t / KW) % KH, xt = t % KW;
                r0o[j] = row_off(mv0, iz0 + zt, iy0 + yt, ix0 + xt, Din, Hin, Win, CIN, zoff_in);
                r1o[j] = row_off(mv1, iz1 + zt, iy1 + yt, ix1 + xt, Din, Hin, Win, CIN, zoff_in);
                wro[j] = wbase + (unsigned)t * wts;
            } else {
                r0o[j] = zoff_in; r1o[j] = zoff_in; wro[j] = zoff_w;
            }
        }
        // A: half0 covers K0..7 (taps 0,1) + K16..23 (taps 4,5);
        //    half1 covers K8..15 (taps 2,3) + K24..31 (taps 6,7).
        v16bf a0 = load_quad4(in + (half ? r0o[2] : r0o[0]), in + (half ? r0o[3] : r0o[1]),
                              in + (half ? r0o[6] : r0o[4]), in + (half ? r0o[7] : r0o[5]));
        v16bf a1 = load_quad4(in + (half ? r1o[2] : r1o[0]), in + (half ? r1o[3] : r1o[1]),
                              in + (half ? r1o[6] : r1o[4]), in + (half ? r1o[7] : r1o[5]));
        // B: half0 = taps 0,1,4,5; half1 = taps 2,3,6,7 (same K as A).
        v16bf b = load_quad4(wsrc + (half ? wro[2] : wro[0]), wsrc + (half ? wro[3] : wro[1]),
                             wsrc + (half ? wro[6] : wro[4]), wsrc + (half ? wro[7] : wro[5]));
        acc0 = wmma_bf16(a0, b, acc0);
        acc1 = wmma_bf16(a1, b, acc1);
    }
    CONV_EPILOGUE()
}

// ---------- final: (D=2,H=40,W=36,C=128) bf16 grid -> (256,40,36) fp32 ----------
__global__ void finalize_out(const u16* __restrict__ grid, float* __restrict__ out) {
    int i = blockIdx.x * blockDim.x + threadIdx.x;
    const int TOT = 256 * 40 * 36;
    if (i >= TOT) return;
    int c2 = i / (40 * 36);
    int r  = i % (40 * 36);
    int h = r / 36, w = r % 36;
    int c = c2 >> 1, d = c2 & 1;
    out[i] = bf2f(grid[(size_t)((d * 40 + h) * 36 + w) * 128 + c]);
}

// ---------- host side ----------
struct LCfg { int cin, cout, kd, kh, kw, sd, sh, sw, pd, ph, pw, sp; };
static const LCfg LYR[12] = {
    {  4,  16, 3,3,3, 1,1,1, 1,1,1, 0 },
    { 16,  16, 3,3,3, 1,1,1, 1,1,1, 0 },
    { 16,  32, 3,3,3, 2,2,2, 1,1,1, 1 },
    { 32,  32, 3,3,3, 1,1,1, 1,1,1, 0 },
    { 32,  32, 3,3,3, 1,1,1, 1,1,1, 0 },
    { 32,  64, 3,3,3, 2,2,2, 1,1,1, 1 },
    { 64,  64, 3,3,3, 1,1,1, 1,1,1, 0 },
    { 64,  64, 3,3,3, 1,1,1, 1,1,1, 0 },
    { 64,  64, 3,3,3, 2,2,2, 0,1,1, 1 },
    { 64,  64, 3,3,3, 1,1,1, 1,1,1, 0 },
    { 64,  64, 3,3,3, 1,1,1, 1,1,1, 0 },
    { 64, 128, 3,1,1, 2,1,1, 0,0,0, 1 },
};
static inline int odim(int in, int k, int s, int p) { return (in + 2 * p - k) / s + 1; }

extern "C" void kernel_launch(void* const* d_in, const int* in_sizes, int n_in,
                              void* d_out, int out_size, void* d_ws, size_t ws_size,
                              hipStream_t stream) {
    (void)in_sizes; (void)n_in; (void)out_size; (void)ws_size;

    // workspace bump allocator (256B aligned)
    char* wp = (char*)d_ws;
    auto alloc = [&](size_t bytes) -> char* {
        char* r = wp;
        wp += (bytes + 255) & ~(size_t)255;
        return r;
    };

    // grids with a zeroed 256-element tail (OOB gather target at offset GRID_ELEMS)
    const size_t GRID_ELEMS = 67108864;   // 64Mi bf16 elems; max layer grid is 60.5M
    const unsigned ZOFF_IN = (unsigned)GRID_ELEMS;
    u16* gA = (u16*)alloc((GRID_ELEMS + 256) * 2);
    u16* gB = (u16*)alloc((GRID_ELEMS + 256) * 2);

    // mask buffers for the 5 resolutions
    unsigned char* masks[5];
    {
        int d = 41, h = 320, w = 288, li = 0;
        masks[li++] = (unsigned char*)alloc((size_t)d * h * w);
        for (int i = 0; i < 12; ++i) {
            int od = odim(d, LYR[i].kd, LYR[i].sd, LYR[i].pd);
            int oh = odim(h, LYR[i].kh, LYR[i].sh, LYR[i].ph);
            int ow = odim(w, LYR[i].kw, LYR[i].sw, LYR[i].pw);
            if (LYR[i].sp) masks[li++] = (unsigned char*)alloc((size_t)od * oh * ow);
            d = od; h = oh; w = ow;
        }
    }

    // bf16 weight region (each layer block + zeroed 256-elem tail)
    u16* wptr[12];
    unsigned wz[12];
    unsigned wcount[12];   // elements staged per layer (block + tail)
    size_t wtot = 0;
    for (int i = 0; i < 12; ++i) {
        size_t el = (size_t)LYR[i].kd * LYR[i].kh * LYR[i].kw * LYR[i].cout * LYR[i].cin;
        wz[i] = (unsigned)el;            // zero tail offset within this layer's block
        wcount[i] = (unsigned)(el + 256);
        wtot += ((el + 256 + 127) & ~(size_t)127);
    }
    u16* wregion = (u16*)alloc(wtot * 2);
    {
        size_t off = 0;
        for (int i = 0; i < 12; ++i) {
            size_t el = (size_t)LYR[i].kd * LYR[i].kh * LYR[i].kw * LYR[i].cout * LYR[i].cin;
            wptr[i] = wregion + off;
            off += ((el + 256 + 127) & ~(size_t)127);
        }
    }
    float* bnbuf = (float*)alloc((size_t)12 * 256 * 4);

    // ---- prep ----
    {
        size_t nz = (size_t)41 * 320 * 288 * 4;
        fill_u16<<<4096, 256, 0, stream>>>(gA, nz);
        fill_u16<<<1, 256, 0, stream>>>(gA + GRID_ELEMS, 256);   // zero tails
        fill_u16<<<1, 256, 0, stream>>>(gB + GRID_ELEMS, 256);
        fill_u16<<<64, 256, 0, stream>>>(wregion, wtot);         // zero whole weight region
        fill_u8<<<4096, 256, 0, stream>>>(masks[0], (size_t)41 * 320 * 288);
        scatter_voxels<<<(20000 + 255) / 256, 256, 0, stream>>>(
            (const float*)d_in[0], (const int*)d_in[1], gA, masks[0], 20000, 320, 288, 4);
    }
    for (int i = 0; i < 12; ++i) {
        const LCfg& L = LYR[i];
        int kt = L.kd * L.kh * L.kw;
        int tot = kt * L.cout * L.cin;
        wprep<<<(tot + 255) / 256, 256, 0, stream>>>(
            (const float*)d_in[2 + 5 * i + 0], wptr[i], kt, L.cout, L.cin);
        bnprep<<<(L.cout + 63) / 64, 64, 0, stream>>>(
            (const float*)d_in[2 + 5 * i + 1], (const float*)d_in[2 + 5 * i + 2],
            (const float*)d_in[2 + 5 * i + 3], (const float*)d_in[2 + 5 * i + 4],
            bnbuf + (size_t)i * 256, L.cout);
    }

    // ---- 12-layer pipeline ----
    u16* gin = gA;
    u16* gout = gB;
    unsigned char* cmask = masks[0];
    int lvl = 0;
    int d = 41, h = 320, w = 288;
    for (int i = 0; i < 12; ++i) {
        const LCfg& L = LYR[i];
        int od = odim(d, L.kd, L.sd, L.pd);
        int oh = odim(h, L.kh, L.sh, L.ph);
        int ow = odim(w, L.kw, L.sw, L.pw);
        int Mo = od * oh * ow;

        unsigned char* omask = cmask;
        if (L.sp) {
            omask = masks[lvl + 1];
            mask_update<<<(Mo + 255) / 256, 256, 0, stream>>>(
                cmask, omask, d, h, w, od, oh, ow,
                L.kd, L.kh, L.kw, L.sd, L.sh, L.sw, L.pd, L.ph, L.pw);
            lvl++;
        }

        int waves = ((Mo + 31) / 32) * (L.cout / 16);
        int blocks = (waves + 7) / 8;
        float* bnp = bnbuf + (size_t)i * 256;
        size_t smb = (size_t)wcount[i] * 2;   // LDS bytes when staging
        if (L.cin == 4) {
            conv_c4<3, 3, 3, true><<<blocks, 256, smb, stream>>>(
                gin, gout, wptr[i], bnp, omask, ZOFF_IN, wz[i], wcount[i],
                d, h, w, od, oh, ow, L.cout, L.sd, L.sh, L.sw, L.pd, L.ph, L.pw);
        } else if (L.cin == 16) {
            conv_c16<3, 3, 3, true><<<blocks, 256, smb, stream>>>(
                gin, gout, wptr[i], bnp, omask, ZOFF_IN, wz[i], wcount[i],
                d, h, w, od, oh, ow, L.cout, L.sd, L.sh, L.sw, L.pd, L.ph, L.pw);
        } else if (L.cin == 32) {
            conv_big<32, 3, 3, 3, true><<<blocks, 256, smb, stream>>>(
                gin, gout, wptr[i], bnp, omask, ZOFF_IN, wcount[i],
                d, h, w, od, oh, ow, L.cout, L.sd, L.sh, L.sw, L.pd, L.ph, L.pw);
        } else if (L.kh == 1) {
            // 3x1x1, cin=64: weight block is 48KB -> stage in LDS
            conv_big<64, 3, 1, 1, true><<<blocks, 256, smb, stream>>>(
                gin, gout, wptr[i], bnp, omask, ZOFF_IN, wcount[i],
                d, h, w, od, oh, ow, L.cout, L.sd, L.sh, L.sw, L.pd, L.ph, L.pw);
        } else {
            // cin=64 3x3x3: 216KB weight block -> keep global/L2 path
            conv_big<64, 3, 3, 3, false><<<blocks, 256, 0, stream>>>(
                gin, gout, wptr[i], bnp, omask, ZOFF_IN, 0,
                d, h, w, od, oh, ow, L.cout, L.sd, L.sh, L.sw, L.pd, L.ph, L.pw);
        }

        cmask = omask;
        d = od; h = oh; w = ow;
        u16* t = gin; gin = gout; gout = t;
    }

    finalize_out<<<(256 * 40 * 36 + 255) / 256, 256, 0, stream>>>(gin, (float*)d_out);
}
